// ImprovedModalGenerator_82935818486189
// MI455X (gfx1250) — matrix operand order repaired and checked by reading, not compiled, not run on hardware
//
#include <hip/hip_runtime.h>

// =====================================================================
// CrossModalGenerator pipeline on MI455X (gfx1250), wave32 + bf16 WMMA.
// ~2.5 TFLOPs of dense GEMM vs ~0.5 GB traffic @23.3TB/s -> compute
// bound -> all linears via v_wmma_f32_16x16x32_bf16.  Weights are
// converted to bf16 once per call; B tiles then stream to LDS with
// gfx1250 async loads (global_load_async_to_lds_b128 / ASYNCcnt).
// =====================================================================

#define H 1024
#define HEADS 16
#define BATCH 8192
#define TM 128
#define TN 64
#define TK 64

enum { ACT_NONE = 0, ACT_GELU = 1, ACT_RELU = 2 };

typedef __attribute__((ext_vector_type(16))) __bf16 v16bf;
typedef __attribute__((ext_vector_type(8)))  float  v8f;
typedef __attribute__((ext_vector_type(8)))  unsigned short u16x8;
typedef __attribute__((ext_vector_type(4)))  unsigned short u16x4;

union FragAB { v16bf v; u16x8 h[2]; };

static __device__ __forceinline__ unsigned short f32_to_bf16(float f) {
  unsigned int u = __float_as_uint(f);
  u += 0x7FFFu + ((u >> 16) & 1u);            // round-to-nearest-even
  return (unsigned short)(u >> 16);
}

static __device__ __forceinline__ float apply_act(float x, int act) {
  if (act == ACT_GELU) return 0.5f * x * (1.0f + erff(x * 0.70710678118654752f));
  if (act == ACT_RELU) return fmaxf(x, 0.0f);
  return x;
}

// f32 -> bf16 weight conversion (once per call; n multiple of 1024)
__global__ __launch_bounds__(256)
void f2bf_kernel(const float* __restrict__ src, unsigned short* __restrict__ dst)
{
  const size_t idx = ((size_t)blockIdx.x * 256 + threadIdx.x) * 4;
  const float4 s = *(const float4*)(src + idx);
  u16x4 d;
  d[0] = f32_to_bf16(s.x); d[1] = f32_to_bf16(s.y);
  d[2] = f32_to_bf16(s.z); d[3] = f32_to_bf16(s.w);
  *(u16x4*)(dst + idx) = d;
}

// ---------------------------------------------------------------------
// C[M,N] = act(A[M,K] (f32, row stride lda) * Wb[N,K]^T (bf16) + bias)
// Block: 256 thr = 8 waves; tile 128x64x64; wave = 32x32 sub-tile.
// A staged via f32->bf16 register convert; B staged with gfx1250 async
// global->LDS copies (ASYNCcnt + s_wait_asynccnt).
// Fragment layouts per CDNA5 ISA 7.12.2.
// ---------------------------------------------------------------------
__global__ __launch_bounds__(256)
void wmma_gemm_bias_act(const float* __restrict__ A, int lda,
                        const unsigned short* __restrict__ Wb,
                        const float* __restrict__ bias,
                        float* __restrict__ C,
                        int M, int N, int K, int act)
{
  __shared__ unsigned short sA[TM][TK + 8];   // +8 pad: rows stay 16B aligned
  __shared__ unsigned short sB[TN][TK + 8];

  const int  t    = threadIdx.x;
  const int  lane = t & 31;
  const int  wid  = t >> 5;
  const int  mb   = (wid & 3) * 32;           // wave M offset inside tile
  const int  nb   = (wid >> 2) * 32;          // wave N offset inside tile
  const long m0   = (long)blockIdx.y * TM;
  const long n0   = (long)blockIdx.x * TN;
  const int  r    = lane & 15;
  const int  hi   = lane >> 4;

  v8f acc[2][2];
  const v8f zero = {0.f,0.f,0.f,0.f,0.f,0.f,0.f,0.f};
  acc[0][0] = zero; acc[0][1] = zero; acc[1][0] = zero; acc[1][1] = zero;

  const unsigned sbBase = (unsigned)(size_t)(&sB[0][0]);

  for (int kk = 0; kk < K; kk += TK) {
    __syncthreads();
    // ---- stage B tile (64 x 64 bf16) via async global->LDS, 16B/lane x2
#pragma unroll
    for (int i = 0; i < 2; ++i) {
      const int idx = t + i * 256;            // 0..511 chunks of 8 bf16
      const int row = idx >> 3;
      const int c8  = idx & 7;
      const unsigned lds = sbBase + (unsigned)(row * (TK + 8) + c8 * 8) * 2u;
      const unsigned long long ga =
          (unsigned long long)(size_t)(Wb + (n0 + row) * (long)K + kk + c8 * 8);
      asm volatile("global_load_async_to_lds_b128 %0, %1, off"
                   :: "v"(lds), "v"(ga) : "memory");
    }
    // ---- stage A tile (128 x 64 f32 -> bf16) : 2048 float4, 8/thread
#pragma unroll
    for (int i = 0; i < 8; ++i) {
      int idx = t + i * 256;
      int row = idx >> 4;                     // 16 float4 per row
      int c4  = idx & 15;
      const float4 s = *(const float4*)(A + (m0 + row) * (long)lda + kk + c4 * 4);
      u16x4 d;
      d[0] = f32_to_bf16(s.x); d[1] = f32_to_bf16(s.y);
      d[2] = f32_to_bf16(s.z); d[3] = f32_to_bf16(s.w);
      *(u16x4*)&sA[row][c4 * 4] = d;
    }
    // ---- prefetch next K tile into cache (global_prefetch_b8)
    if (kk + TK < K) {
      __builtin_prefetch(A + (m0 + (t & 127)) * (long)lda + kk + TK + (t >> 7) * 32, 0, 1);
      if (t < TN) __builtin_prefetch(Wb + (n0 + t) * (long)K + kk + TK, 0, 1);
    }
    asm volatile("s_wait_asynccnt 0" ::: "memory");
    __syncthreads();

#pragma unroll
    for (int ks = 0; ks < TK; ks += 32) {
      FragAB a0, a1, b0, b1;
      // A 16x32 frag: lanes<16 K=[0,8)+[16,24), lanes>=16 K=[8,16)+[24,32)
      const int kh = ks + hi * 8;
      a0.h[0] = *(const u16x8*)&sA[mb + r     ][kh];
      a0.h[1] = *(const u16x8*)&sA[mb + r     ][kh + 16];
      a1.h[0] = *(const u16x8*)&sA[mb + 16 + r][kh];
      a1.h[1] = *(const u16x8*)&sA[mb + 16 + r][kh + 16];
      // B 32x16 frag: lane = N col, lanes<16 K=[0,8)+[16,24), lanes>=16 K=[8,16)+[24,32)
      const int kb = ks + hi * 8;
      b0.h[0] = *(const u16x8*)&sB[nb + r     ][kb];
      b0.h[1] = *(const u16x8*)&sB[nb + r     ][kb + 16];
      b1.h[0] = *(const u16x8*)&sB[nb + 16 + r][kb];
      b1.h[1] = *(const u16x8*)&sB[nb + 16 + r][kb + 16];

      acc[0][0] = __builtin_amdgcn_wmma_f32_16x16x32_bf16(false, a0.v, false, b0.v,
                                                          (short)0, acc[0][0], false, false);
      acc[0][1] = __builtin_amdgcn_wmma_f32_16x16x32_bf16(false, a0.v, false, b1.v,
                                                          (short)0, acc[0][1], false, false);
      acc[1][0] = __builtin_amdgcn_wmma_f32_16x16x32_bf16(false, a1.v, false, b0.v,
                                                          (short)0, acc[1][0], false, false);
      acc[1][1] = __builtin_amdgcn_wmma_f32_16x16x32_bf16(false, a1.v, false, b1.v,
                                                          (short)0, acc[1][1], false, false);
    }
  }

  // ---- epilogue: C/D layout -> lane N = lane&15, VGPR e -> M = e (+8 if lane>=16)
#pragma unroll
  for (int i = 0; i < 2; ++i) {
#pragma unroll
    for (int j = 0; j < 2; ++j) {
      const long gn    = n0 + nb + j * 16 + r;
      const float bv   = bias[gn];
      const long mbase = m0 + mb + i * 16 + hi * 8;
#pragma unroll
      for (int e = 0; e < 8; ++e)
        C[(mbase + e) * (long)N + gn] = apply_act(acc[i][j][e] + bv, act);
    }
  }
}

// ---------------------------------------------------------------------
// Seq-2 MHA core: one wave32 per (sample, head); lane covers dims d,d+32.
// ---------------------------------------------------------------------
__global__ __launch_bounds__(256)
void attn2_kernel(const float* __restrict__ qkv, float* __restrict__ o, int B)
{
  const int lane = threadIdx.x & 31;
  const int w = blockIdx.x * 8 + (threadIdx.x >> 5);
  if (w >= B * HEADS) return;
  const int b = w >> 4, h = w & 15;
  const float* r0 = qkv + (size_t)(2 * b) * (3 * H) + h * 64;
  const float* r1 = r0 + 3 * H;
  const int d0 = lane, d1 = lane + 32;
  const float q00 = r0[d0], q01 = r0[d1], q10 = r1[d0], q11 = r1[d1];
  const float k00 = r0[H + d0], k01 = r0[H + d1], k10 = r1[H + d0], k11 = r1[H + d1];
  const float v00 = r0[2*H + d0], v01 = r0[2*H + d1], v10 = r1[2*H + d0], v11 = r1[2*H + d1];
  float s00 = q00*k00 + q01*k01, s01 = q00*k10 + q01*k11;
  float s10 = q10*k00 + q11*k01, s11 = q10*k10 + q11*k11;
#pragma unroll
  for (int m = 16; m >= 1; m >>= 1) {
    s00 += __shfl_xor(s00, m, 32);
    s01 += __shfl_xor(s01, m, 32);
    s10 += __shfl_xor(s10, m, 32);
    s11 += __shfl_xor(s11, m, 32);
  }
  s00 *= 0.125f; s01 *= 0.125f; s10 *= 0.125f; s11 *= 0.125f;   // 1/sqrt(64)
  const float m0v = fmaxf(s00, s01), m1v = fmaxf(s10, s11);
  const float e00 = __expf(s00 - m0v), e01 = __expf(s01 - m0v);
  const float e10 = __expf(s10 - m1v), e11 = __expf(s11 - m1v);
  const float a00 = e00 / (e00 + e01), a01 = e01 / (e00 + e01);
  const float a10 = e10 / (e10 + e11), a11 = e11 / (e10 + e11);
  float* o0 = o + (size_t)(2 * b) * H + h * 64;
  float* o1 = o0 + H;
  o0[d0] = a00 * v00 + a01 * v10;  o0[d1] = a00 * v01 + a01 * v11;
  o1[d0] = a10 * v00 + a11 * v10;  o1[d1] = a10 * v01 + a11 * v11;
}

// x = LayerNorm(x + res) * g + b, one block per row of H=1024.
__global__ __launch_bounds__(256)
void ln_res_kernel(float* __restrict__ x, const float* __restrict__ res,
                   const float* __restrict__ g, const float* __restrict__ bta)
{
  __shared__ float rs[256], rq[256];
  float* xr = x + (size_t)blockIdx.x * H;
  const float* rr = res + (size_t)blockIdx.x * H;
  const int t = threadIdx.x;
  float v[4], s = 0.f, s2 = 0.f;
#pragma unroll
  for (int i = 0; i < 4; ++i) {
    const int idx = t + i * 256;
    const float u = xr[idx] + rr[idx];
    v[i] = u; s += u; s2 += u * u;
  }
  rs[t] = s; rq[t] = s2; __syncthreads();
  for (int o = 128; o > 0; o >>= 1) {
    if (t < o) { rs[t] += rs[t + o]; rq[t] += rq[t + o]; }
    __syncthreads();
  }
  const float mean = rs[0] * (1.f / H);
  const float var  = rq[0] * (1.f / H) - mean * mean;
  const float inv  = rsqrtf(var + 1e-5f);
#pragma unroll
  for (int i = 0; i < 4; ++i) {
    const int idx = t + i * 256;
    xr[idx] = (v[i] - mean) * inv * g[idx] + bta[idx];
  }
}

// x[2b] = t0[b], x[2b+1] = prior[b]
__global__ __launch_bounds__(256)
void interleave_kernel(const float* __restrict__ t0, const float* __restrict__ prior,
                       float* __restrict__ x)
{
  const size_t idx = (size_t)blockIdx.x * 256 + threadIdx.x;
  const size_t b = idx >> 10, d = idx & 1023;
  x[(2 * b) * H + d]     = t0[idx];
  x[(2 * b + 1) * H + d] = prior[idx];
}

// out = rw*prior + (1-rw)*tanh(g0)
__global__ __launch_bounds__(256)
void mixtanh_kernel(const float* __restrict__ g0, const float* __restrict__ prior,
                    const float* __restrict__ rw, float* __restrict__ out)
{
  const size_t idx = (size_t)blockIdx.x * 256 + threadIdx.x;
  const float r = rw[0];
  out[idx] = r * prior[idx] + (1.0f - r) * tanhf(g0[idx]);
}

// cat[b, 0:H] = feat[b], cat[b, H:2H] = gen[b]
__global__ __launch_bounds__(256)
void concat_kernel(const float* __restrict__ feat, const float* __restrict__ gen,
                   float* __restrict__ cat)
{
  const size_t idx = (size_t)blockIdx.x * 256 + threadIdx.x;
  const size_t b = idx >> 10, d = idx & 1023;
  cat[b * (2 * H) + d]     = feat[idx];
  cat[b * (2 * H) + H + d] = gen[idx];
}

// q[b] = sigmoid(dot(h[b,:], w2) + b2); one wave per row.
__global__ __launch_bounds__(256)
void qual_out_kernel(const float* __restrict__ hbuf, const float* __restrict__ w2,
                     const float* __restrict__ b2, float* __restrict__ q, int B)
{
  const int lane = threadIdx.x & 31;
  const int b = blockIdx.x * 8 + (threadIdx.x >> 5);
  if (b >= B) return;
  const float* hr = hbuf + (size_t)b * H;
  float s = 0.f;
#pragma unroll
  for (int d = 0; d < H; d += 32) s += hr[d + lane] * w2[d + lane];
#pragma unroll
  for (int m = 16; m >= 1; m >>= 1) s += __shfl_xor(s, m, 32);
  if (lane == 0) q[b] = 1.0f / (1.0f + expf(-(s + b2[0])));
}

__global__ __launch_bounds__(256)
void select_feat_kernel(const float* __restrict__ img, const float* __restrict__ txt,
                        const float* __restrict__ gtex, const float* __restrict__ gimg,
                        const int* __restrict__ mt, float* __restrict__ out)
{
  const size_t idx = (size_t)blockIdx.x * 256 + threadIdx.x;
  const int m = mt[idx >> 10];
  out[idx]                      = (m == 2) ? gimg[idx] : img[idx];   // enhanced_img
  out[(size_t)BATCH * H + idx]  = (m == 1) ? gtex[idx] : txt[idx];   // enhanced_text
}

__global__ __launch_bounds__(256)
void select_small_kernel(const int* __restrict__ mt, const float* __restrict__ qtex,
                         const float* __restrict__ qimg, float* __restrict__ mask,
                         float* __restrict__ qual)
{
  const int b = blockIdx.x * 256 + threadIdx.x;
  const bool m1 = (mt[b] == 1), m2 = (mt[b] == 2);
  mask[2 * b]     = m2 ? 1.0f : 0.0f;
  mask[2 * b + 1] = m1 ? 1.0f : 0.0f;
  qual[2 * b]     = m2 ? qimg[b] : 1.0f;
  qual[2 * b + 1] = m1 ? qtex[b] : 1.0f;
}

// =====================================================================
// Host orchestration
// =====================================================================
struct GenParams {
  const float *in_w, *in_b, *o1_w, *o1_b, *o2_w, *o2_b, *rw;
  const float *w_in[3], *b_in[3], *w_out[3], *b_out[3];
  const float *g1[3], *b1[3], *w1[3], *bb1[3], *w2[3], *bb2[3], *g2[3], *b2[3];
};
struct GenWB {   // bf16 copies of the weight matrices
  const unsigned short *in_w, *o1_w, *o2_w;
  const unsigned short *w_in[3], *w_out[3], *w1[3], *w2[3];
};

static void fill_gen_sorted(GenParams& P, void* const* d, int g) {
  // jax tree_flatten order: dict keys sorted at every level
  for (int j = 0; j < 3; ++j) {
    const int b = g + j * 12;
    P.b_in[j]  = (const float*)d[b + 0];  P.b_out[j] = (const float*)d[b + 1];
    P.w_in[j]  = (const float*)d[b + 2];  P.w_out[j] = (const float*)d[b + 3];
    P.b1[j]    = (const float*)d[b + 4];  P.b2[j]    = (const float*)d[b + 5];
    P.bb1[j]   = (const float*)d[b + 6];  P.bb2[j]   = (const float*)d[b + 7];
    P.g1[j]    = (const float*)d[b + 8];  P.g2[j]    = (const float*)d[b + 9];
    P.w1[j]    = (const float*)d[b + 10]; P.w2[j]    = (const float*)d[b + 11];
  }
  P.in_b = (const float*)d[g + 36]; P.in_w = (const float*)d[g + 37];
  P.o1_b = (const float*)d[g + 38]; P.o1_w = (const float*)d[g + 39];
  P.o2_b = (const float*)d[g + 40]; P.o2_w = (const float*)d[g + 41];
  P.rw   = (const float*)d[g + 42];
}

static void fill_gen_insertion(GenParams& P, void* const* d, int g) {
  P.in_w = (const float*)d[g + 0]; P.in_b = (const float*)d[g + 1];
  for (int j = 0; j < 3; ++j) {
    const int b = g + 2 + j * 12;
    P.w_in[j]  = (const float*)d[b + 0];  P.b_in[j]  = (const float*)d[b + 1];
    P.w_out[j] = (const float*)d[b + 2];  P.b_out[j] = (const float*)d[b + 3];
    P.g1[j]    = (const float*)d[b + 4];  P.b1[j]    = (const float*)d[b + 5];
    P.w1[j]    = (const float*)d[b + 6];  P.bb1[j]   = (const float*)d[b + 7];
    P.w2[j]    = (const float*)d[b + 8];  P.bb2[j]   = (const float*)d[b + 9];
    P.g2[j]    = (const float*)d[b + 10]; P.b2[j]    = (const float*)d[b + 11];
  }
  P.o1_w = (const float*)d[g + 38]; P.o1_b = (const float*)d[g + 39];
  P.o2_w = (const float*)d[g + 40]; P.o2_b = (const float*)d[g + 41];
  P.rw   = (const float*)d[g + 42];
}

static const unsigned short* convw(hipStream_t s, const float* src, size_t n,
                                   unsigned short*& pool) {
  unsigned short* dst = pool; pool += n;
  f2bf_kernel<<<(unsigned)(n / 1024), 256, 0, s>>>(src, dst);
  return dst;
}

static void conv_gen(hipStream_t s, const GenParams& P, GenWB& W, unsigned short*& pool) {
  W.in_w = convw(s, P.in_w, (size_t)H * H, pool);
  for (int j = 0; j < 3; ++j) {
    W.w_in[j]  = convw(s, P.w_in[j],  (size_t)3 * H * H, pool);
    W.w_out[j] = convw(s, P.w_out[j], (size_t)H * H, pool);
    W.w1[j]    = convw(s, P.w1[j],    (size_t)4 * H * H, pool);
    W.w2[j]    = convw(s, P.w2[j],    (size_t)4 * H * H, pool);
  }
  W.o1_w = convw(s, P.o1_w, (size_t)2 * H * H, pool);
  W.o2_w = convw(s, P.o2_w, (size_t)2 * H * H, pool);
}

static inline void gemm(hipStream_t s, const float* A, int lda, const unsigned short* Wb,
                        const float* bias, float* C, int M, int N, int K, int act) {
  dim3 grid(N / TN, M / TM);
  wmma_gemm_bias_act<<<grid, 256, 0, s>>>(A, lda, Wb, bias, C, M, N, K, act);
}

static void run_gen(hipStream_t s, const GenParams& P, const GenWB& W, const float* src,
                    const float* prior, float* gen_out,
                    float* xb, float* big, float* attno, float* ffo,
                    float* tmpB, float* head2)
{
  const int M2 = 2 * BATCH;
  gemm(s, src, H, W.in_w, P.in_b, tmpB, BATCH, H, H, ACT_NONE);
  interleave_kernel<<<(BATCH * H) / 256, 256, 0, s>>>(tmpB, prior, xb);
  for (int j = 0; j < 3; ++j) {
    gemm(s, xb, H, W.w_in[j], P.b_in[j], big, M2, 3 * H, H, ACT_NONE);
    attn2_kernel<<<(BATCH * HEADS) / 8, 256, 0, s>>>(big, attno, BATCH);
    gemm(s, attno, H, W.w_out[j], P.b_out[j], ffo, M2, H, H, ACT_NONE);
    ln_res_kernel<<<M2, 256, 0, s>>>(xb, ffo, P.g1[j], P.b1[j]);
    gemm(s, xb, H, W.w1[j], P.bb1[j], big, M2, 4 * H, H, ACT_GELU);
    gemm(s, big, 4 * H, W.w2[j], P.bb2[j], ffo, M2, H, 4 * H, ACT_NONE);
    ln_res_kernel<<<M2, 256, 0, s>>>(xb, ffo, P.g2[j], P.b2[j]);
  }
  // head reads token 0 only -> A = xb with row stride 2H (even rows)
  gemm(s, xb, 2 * H, W.o1_w, P.o1_b, head2, BATCH, 2 * H, H, ACT_GELU);
  gemm(s, head2, 2 * H, W.o2_w, P.o2_b, tmpB, BATCH, H, 2 * H, ACT_NONE);
  mixtanh_kernel<<<(BATCH * H) / 256, 256, 0, s>>>(tmpB, prior, P.rw, gen_out);
}

extern "C" void kernel_launch(void* const* d_in, const int* in_sizes, int n_in,
                              void* d_out, int out_size, void* d_ws, size_t ws_size,
                              hipStream_t stream)
{
  (void)n_in; (void)out_size; (void)ws_size;
  // Detect pytree flattening order: jax tree_flatten sorts dict keys
  // (-> in_sizes[1] == BATCH for missing_type); plain insertion order
  // puts text_features (B*H) second.
  const bool sorted = (in_sizes[1] == BATCH);

  const float *img, *txt; const int* mt;
  const float *qw1, *qb1, *qw2, *qb2;
  GenParams Pi2t, Pt2i;
  if (sorted) {
    img = (const float*)d_in[0];
    mt  = (const int*)d_in[1];
    fill_gen_sorted(Pi2t, d_in, 2);
    qb1 = (const float*)d_in[45]; qb2 = (const float*)d_in[46];
    qw1 = (const float*)d_in[47]; qw2 = (const float*)d_in[48];
    fill_gen_sorted(Pt2i, d_in, 49);
    txt = (const float*)d_in[92];
  } else {
    img = (const float*)d_in[0];
    txt = (const float*)d_in[1];
    mt  = (const int*)d_in[2];
    fill_gen_insertion(Pi2t, d_in, 3);
    fill_gen_insertion(Pt2i, d_in, 46);
    qw1 = (const float*)d_in[89]; qb1 = (const float*)d_in[90];
    qw2 = (const float*)d_in[91]; qb2 = (const float*)d_in[92];
  }

  // ---- carve workspace (f32 activations + bf16 weight pool)
  float* ws = (float*)d_ws;
  size_t off = 0;
  auto carve = [&](size_t n) { float* p = ws + off; off += n; return p; };
  float* xb    = carve((size_t)2 * BATCH * H);        // [2B, H]
  float* big   = carve((size_t)2 * BATCH * 4 * H);    // [2B, 4H] (qkv & ffn hidden)
  float* attno = carve((size_t)2 * BATCH * H);        // [2B, H]
  float* ffo   = carve((size_t)2 * BATCH * H);        // [2B, H]
  float* tmpB  = carve((size_t)BATCH * H);            // [B, H]
  float* head2 = carve((size_t)BATCH * 2 * H);        // [B, 2H] (head & concat)
  float* gtex  = carve((size_t)BATCH * H);
  float* gimg  = carve((size_t)BATCH * H);
  float* qtex  = carve(BATCH);
  float* qimg  = carve(BATCH);
  unsigned short* pool = (unsigned short*)carve(((size_t)84 * 1024 * 1024 + 2 * H * H) / 2);

  // one-time (per call) f32 -> bf16 weight conversion
  GenWB Wi2t, Wt2i;
  conv_gen(stream, Pi2t, Wi2t, pool);
  conv_gen(stream, Pt2i, Wt2i, pool);
  const unsigned short* qw1b = convw(stream, qw1, (size_t)2 * H * H, pool);

  // gen_text = gen(image, prior=text, p_i2t); gen_img = gen(text, prior=image, p_t2i)
  run_gen(stream, Pi2t, Wi2t, img, txt, gtex, xb, big, attno, ffo, tmpB, head2);
  run_gen(stream, Pt2i, Wt2i, txt, img, gimg, xb, big, attno, ffo, tmpB, head2);

  // quality(concat(feat, gen))
  concat_kernel<<<(BATCH * H) / 256, 256, 0, stream>>>(img, gtex, head2);
  gemm(stream, head2, 2 * H, qw1b, qb1, tmpB, BATCH, H, 2 * H, ACT_RELU);
  qual_out_kernel<<<BATCH / 8, 256, 0, stream>>>(tmpB, qw2, qb2, qtex, BATCH);

  concat_kernel<<<(BATCH * H) / 256, 256, 0, stream>>>(txt, gimg, head2);
  gemm(stream, head2, 2 * H, qw1b, qb1, tmpB, BATCH, H, 2 * H, ACT_RELU);
  qual_out_kernel<<<BATCH / 8, 256, 0, stream>>>(tmpB, qw2, qb2, qimg, BATCH);

  // outputs: [enhanced_img | enhanced_text | mask(B,2) | quality(B,2)]
  float* out = (float*)d_out;
  select_feat_kernel<<<(BATCH * H) / 256, 256, 0, stream>>>(img, txt, gtex, gimg, mt, out);
  select_small_kernel<<<BATCH / 256, 256, 0, stream>>>(
      mt, qtex, qimg, out + (size_t)2 * BATCH * H, out + (size_t)2 * BATCH * H + 2 * BATCH);
}
